// PL_MSA_8727373545598
// MI455X (gfx1250) — matrix operand order repaired and checked by reading, not compiled, not run on hardware
//
#include <hip/hip_runtime.h>

// ---------------------------------------------------------------------------
// Fused windowed attention for MI455X (gfx1250, wave32, WMMA f16->f32).
//
// DIM=180 HEADS=6 HEAD_DIM=30 N=256 M=64 BW=1024 windows.
// One workgroup (256 thr = 8 waves) per window; all intermediates in LDS.
// ---------------------------------------------------------------------------

typedef _Float16 v16h __attribute__((ext_vector_type(16)));
typedef _Float16 h8   __attribute__((ext_vector_type(8)));
typedef float    v8f  __attribute__((ext_vector_type(8)));

#define KPAD 192              // 180 padded to 6 x 32
#define SCALE 0.18257418583505536f   // 30^-0.5

// LDS byte offsets (total 311,296 B dynamic LDS)
#define OFF_XM   0            // [256][192] f16  (phase 1-2: xm; phase 3-4: ao)
#define OFF_Q    98304        // [6][256][32] f16
#define OFF_K    196608       // [6][64][32]  f16  (m-major: k[m][d])
#define OFF_V    221184       // [6][32][64]  f16  (d-major: v[d][m])
#define OFF_AT   245760       // [256][64]    f32
#define LDS_BYTES 311296

__device__ __forceinline__ v8f wmma16(v16h a, v16h b, v8f c) {
  // emits v_wmma_f32_16x16x32_f16
  return __builtin_amdgcn_wmma_f32_16x16x32_f16(
      /*neg_a=*/false, a, /*neg_b=*/false, b,
      /*c_mod=*/(short)0, c, /*reuse_a=*/false, /*reuse_b=*/false);
}

// A fragment (16x32 f16): row fixed per lane; K = (i&7) + 16*(i>>3) + 8*hb
// -> two contiguous runs of 8 halves at +8*hb and +16+8*hb.
__device__ __forceinline__ v16h frag_a(const _Float16* p, int hb) {
  h8 lo = *(const h8*)(p + 8 * hb);
  h8 hi = *(const h8*)(p + 16 + 8 * hb);
  v16h r;
#pragma unroll
  for (int i = 0; i < 8; ++i) { r[i] = lo[i]; r[i + 8] = hi[i]; }
  return r;
}

// A fragment built from an fp32 row (post-softmax attention).
__device__ __forceinline__ v16h frag_a_f32(const float* p, int hb) {
  v16h r;
#pragma unroll
  for (int i = 0; i < 8; ++i) {
    r[i]     = (_Float16)p[8 * hb + i];
    r[i + 8] = (_Float16)p[16 + 8 * hb + i];
  }
  return r;
}

// B fragment (32x16 f16): col fixed per lane; K = (i&7) + 16*(i>>3) + 8*hb.
__device__ __forceinline__ v16h frag_b(const _Float16* p, int hb) {
  h8 lo = *(const h8*)(p + 8 * hb);
  h8 hi = *(const h8*)(p + 16 + 8 * hb);
  v16h r;
#pragma unroll
  for (int i = 0; i < 8; ++i) { r[i] = lo[i]; r[i + 8] = hi[i]; }
  return r;
}

// --------------------------- tiny prep kernels -----------------------------

// film[img][cc] = sum_i k_v[img][i] * W_kernel[cc][i]   (4 x 360)
__global__ void film_kernel(const float* __restrict__ k_v,
                            const float* __restrict__ Wk,
                            float* __restrict__ film) {
  int idx = blockIdx.x * blockDim.x + threadIdx.x;
  if (idx >= 4 * 360) return;
  int im = idx / 360, cc = idx - im * 360;
  const float* a = k_v + im * 256;
  const float* w = Wk + cc * 256;
  float s = 0.f;
#pragma unroll 8
  for (int i = 0; i < 256; ++i) s += a[i] * w[i];
  film[idx] = s;
}

// f16 weight staging, zero padded: Wq 192x192, Wkv 96x192, Wp 192x192
__global__ void wconv_kernel(const float* __restrict__ Wq,
                             const float* __restrict__ Wkv,
                             const float* __restrict__ Wp,
                             _Float16* __restrict__ Wqh,
                             _Float16* __restrict__ Wkvh,
                             _Float16* __restrict__ Wph) {
  int i = blockIdx.x * blockDim.x + threadIdx.x;
  if (i >= 92160) return;
  if (i < 36864) {
    int o = i / KPAD, c = i - o * KPAD;
    Wqh[i] = (o < 180 && c < 180) ? (_Float16)Wq[o * 180 + c] : (_Float16)0.f;
  } else if (i < 55296) {
    int j = i - 36864;
    int o = j / KPAD, c = j - o * KPAD;
    Wkvh[j] = (o < 90 && c < 180) ? (_Float16)Wkv[o * 180 + c] : (_Float16)0.f;
  } else {
    int j = i - 55296;
    int o = j / KPAD, c = j - o * KPAD;
    Wph[j] = (o < 180 && c < 180) ? (_Float16)Wp[o * 180 + c] : (_Float16)0.f;
  }
}

// --------------------------- fused main kernel -----------------------------

__global__ __launch_bounds__(256) void fused_window_attn(
    const float* __restrict__ x,          // (1024,256,180)
    const float* __restrict__ mask,       // (256,256,64)
    const float* __restrict__ b_q,        // (180)
    const float* __restrict__ b_kv,       // (90)
    const float* __restrict__ bias_table, // (225,6)
    const float* __restrict__ b_proj,     // (180)
    const int*   __restrict__ rel_index,  // (256,64)
    const float* __restrict__ film,       // (4,360)
    const _Float16* __restrict__ Wqh,     // 192x192
    const _Float16* __restrict__ Wkvh,    // 96x192
    const _Float16* __restrict__ Wph,     // 192x192
    float* __restrict__ out)              // (1024,256,180)
{
  extern __shared__ char smem[];
  _Float16* xm = (_Float16*)(smem + OFF_XM);  // later reused as "ao"
  _Float16* qs = (_Float16*)(smem + OFF_Q);
  _Float16* ks = (_Float16*)(smem + OFF_K);
  _Float16* vs = (_Float16*)(smem + OFF_V);
  float*    at = (float*)(smem + OFF_AT);

  const int tid  = threadIdx.x;
  const int wave = tid >> 5;
  const int lane = tid & 31;
  const int lo   = lane & 15;  // A-row / B-col / D-col within tile
  const int hb   = lane >> 4;  // half-wave select
  const int bw   = blockIdx.x; // window index b_
  const int img  = bw >> 8;
  const int win  = bw & 255;

  // Zero q/k/v LDS (covers d=30,31 padding).
  {
    float* z = (float*)(smem + OFF_Q);
    for (int i = tid; i < (98304 + 24576 + 24576) / 4; i += 256) z[i] = 0.f;
  }

  // ---- Phase 1: FiLM modulation -> xm f16 (K padded to 192) ----
  const float* xw = x + (size_t)bw * 256 * 180;
  const float* f1 = film + img * 360;
  const float* f2 = f1 + 180;
  for (int i = tid; i < 256 * KPAD; i += 256) {
    int n = i / KPAD, c = i - n * KPAD;
    float v = 0.f;
    if (c < 180) v = xw[n * 180 + c] * f1[c] + f2[c];
    xm[i] = (_Float16)v;
  }
  __syncthreads();

  // ---- Phase 2a: KV projection (256x192 @ 192x96) + 2x2-pool scatter ----
  for (int t = wave; t < 96; t += 8) {                // 16 mTiles x 6 nTiles
    int mT = t / 6, nT = t - (t / 6) * 6;
    const _Float16* arow = xm + (mT * 16 + lo) * KPAD;
    const _Float16* brow = Wkvh + (nT * 16 + lo) * KPAD;
    v8f acc = {};
#pragma unroll
    for (int kc = 0; kc < 6; ++kc)
      acc = wmma16(frag_a(arow + kc * 32, hb), frag_b(brow + kc * 32, hb), acc);
    int cc = nT * 16 + lo;                            // kv output channel (<96)
    if (cc < 90) {
      float bias = b_kv[cc];
      int s  = cc >= 45;
      int c4 = cc - s * 45;
#pragma unroll
      for (int r = 0; r < 8; ++r) {
        int n = mT * 16 + r + 8 * hb;
        int row = n >> 4, col = n & 15;
        int m    = (row >> 1) * 8 + (col >> 1);
        int flat = (row & 1) * 90 + (col & 1) * 45 + c4;
        int h = flat / 30, d = flat - (flat / 30) * 30;
        float val = acc[r] + bias;
        if (s == 0) ks[h * 2048 + m * 32 + d] = (_Float16)val;  // k[m][d]
        else        vs[h * 2048 + d * 64 + m] = (_Float16)val;  // v[d][m]
      }
    }
  }

  // ---- Phase 2b: Q projection (256x192 @ 192x192), scaled ----
  for (int t = wave; t < 192; t += 8) {               // 16 mTiles x 12 nTiles
    int mT = t / 12, nT = t - (t / 12) * 12;
    const _Float16* arow = xm + (mT * 16 + lo) * KPAD;
    const _Float16* brow = Wqh + (nT * 16 + lo) * KPAD;
    v8f acc = {};
#pragma unroll
    for (int kc = 0; kc < 6; ++kc)
      acc = wmma16(frag_a(arow + kc * 32, hb), frag_b(brow + kc * 32, hb), acc);
    int o = nT * 16 + lo;
    if (o < 180) {
      int h = o / 30, d = o - (o / 30) * 30;
      float bq = b_q[o];
#pragma unroll
      for (int r = 0; r < 8; ++r) {
        int n = mT * 16 + r + 8 * hb;
        qs[h * 8192 + n * 32 + d] = (_Float16)((acc[r] + bq) * SCALE);
      }
    }
  }
  __syncthreads();

  // xm is dead; reuse its region as attention output "ao" -> zero pads.
  for (int i = tid; i < 256 * KPAD / 2; i += 256) ((float*)xm)[i] = 0.f;
  _Float16* ao = xm;

  // ---- Phase 3: per-head attention ----
  const float* maskw = mask + (size_t)win * 256 * 64;
  for (int h = 0; h < 6; ++h) {
    // QK^T: 16 nTiles x 4 mTiles, K = 32 (head_dim padded)
    for (int t = wave; t < 64; t += 8) {
      int mT = t >> 2, nT = t & 3;
      v16h a = frag_a(qs + h * 8192 + (mT * 16 + lo) * 32, hb);
      v16h b = frag_b(ks + h * 2048 + (nT * 16 + lo) * 32, hb);
      v8f acc = {};
      acc = wmma16(a, b, acc);
      int m = nT * 16 + lo;
#pragma unroll
      for (int r = 0; r < 8; ++r) {
        int n = mT * 16 + r + 8 * hb;
        float val = acc[r] + bias_table[rel_index[n * 64 + m] * 6 + h]
                           + maskw[n * 64 + m];
        at[n * 64 + m] = val;
      }
    }
    __syncthreads();

    // Row softmax over 64 (one row per thread), fp32.
    {
      float* rowp = at + tid * 64;
      float mx = -1e30f;
#pragma unroll 8
      for (int j = 0; j < 64; ++j) mx = fmaxf(mx, rowp[j]);
      float s = 0.f;
#pragma unroll 8
      for (int j = 0; j < 64; ++j) { float e = __expf(rowp[j] - mx); rowp[j] = e; s += e; }
      float inv = 1.f / s;
#pragma unroll 8
      for (int j = 0; j < 64; ++j) rowp[j] *= inv;
    }
    __syncthreads();

    // PV: 16 nTiles x 2 dTiles, K = 64 (2 chunks)
    for (int t = wave; t < 32; t += 8) {
      int mT = t >> 1, nT = t & 1;
      const float*    arow = at + (mT * 16 + lo) * 64;
      const _Float16* brow = vs + h * 2048 + (nT * 16 + lo) * 64;
      v8f acc = {};
#pragma unroll
      for (int kc = 0; kc < 2; ++kc)
        acc = wmma16(frag_a_f32(arow + kc * 32, hb), frag_b(brow + kc * 32, hb), acc);
      int d = nT * 16 + lo;
      if (d < 30) {
#pragma unroll
        for (int r = 0; r < 8; ++r) {
          int n = mT * 16 + r + 8 * hb;
          ao[n * KPAD + h * 30 + d] = (_Float16)acc[r];
        }
      }
    }
    __syncthreads();
  }

  // ---- Phase 4: output projection (256x192 @ 192x192) + bias -> d_out ----
  float* outw = out + (size_t)bw * 256 * 180;
  for (int t = wave; t < 192; t += 8) {
    int mT = t / 12, nT = t - (t / 12) * 12;
    const _Float16* arow = ao + (mT * 16 + lo) * KPAD;
    const _Float16* brow = Wph + (nT * 16 + lo) * KPAD;
    v8f acc = {};
#pragma unroll
    for (int kc = 0; kc < 6; ++kc)
      acc = wmma16(frag_a(arow + kc * 32, hb), frag_b(brow + kc * 32, hb), acc);
    int o = nT * 16 + lo;
    if (o < 180) {
      float bp = b_proj[o];
#pragma unroll
      for (int r = 0; r < 8; ++r) {
        int n = mT * 16 + r + 8 * hb;
        outw[n * 180 + o] = acc[r] + bp;
      }
    }
  }
}

// ------------------------------- launcher ----------------------------------

extern "C" void kernel_launch(void* const* d_in, const int* in_sizes, int n_in,
                              void* d_out, int out_size, void* d_ws, size_t ws_size,
                              hipStream_t stream) {
  const float* x          = (const float*)d_in[0];
  const float* k_v        = (const float*)d_in[1];
  const float* mask       = (const float*)d_in[2];
  const float* W_kernel   = (const float*)d_in[3];
  const float* W_q        = (const float*)d_in[4];
  const float* b_q        = (const float*)d_in[5];
  const float* W_kv       = (const float*)d_in[6];
  const float* b_kv       = (const float*)d_in[7];
  const float* bias_table = (const float*)d_in[8];
  const float* W_proj     = (const float*)d_in[9];
  const float* b_proj     = (const float*)d_in[10];
  const int*   rel_index  = (const int*)d_in[11];
  float* out = (float*)d_out;

  char* ws = (char*)d_ws;
  float*    film = (float*)ws;                          // 1440 f32
  _Float16* Wqh  = (_Float16*)(ws + 8192);              // 192*192 f16
  _Float16* Wkvh = (_Float16*)(ws + 8192 + 73728);      //  96*192 f16
  _Float16* Wph  = (_Float16*)(ws + 8192 + 73728 + 36864); // 192*192 f16

  hipFuncSetAttribute((const void*)fused_window_attn,
                      hipFuncAttributeMaxDynamicSharedMemorySize, LDS_BYTES);

  film_kernel<<<6, 256, 0, stream>>>(k_v, W_kernel, film);
  wconv_kernel<<<360, 256, 0, stream>>>(W_q, W_kv, W_proj, Wqh, Wkvh, Wph);
  fused_window_attn<<<1024, 256, LDS_BYTES, stream>>>(
      x, mask, b_q, b_kv, bias_table, b_proj, rel_index,
      film, Wqh, Wkvh, Wph, out);
}
